// CityMoE_13331578487346
// MI455X (gfx1250) — matrix operand order repaired and checked by reading, not compiled, not run on hardware
//
#include <hip/hip_runtime.h>
#include <hip/hip_bf16.h>

// ---------------------------------------------------------------------------
// MoE (top-2 of 8 experts, SwiGLU) + shared expert for MI455X (gfx1250).
// Compute-bound (~103 GFLOP sparse vs ~6us of HBM traffic) => bf16 WMMA path.
// Weights converted once to bf16 in workspace; they fit in the 192MB L2.
// Deterministic sparse dispatch (prefix-scan compaction, no float atomics).
// M-tile = 64 rows/block to halve L2 weight re-streaming (the secondary
// bottleneck after WMMA throughput).
// ---------------------------------------------------------------------------

typedef unsigned short ushort_t;
typedef __attribute__((ext_vector_type(16))) __bf16 v16bf;
typedef __attribute__((ext_vector_type(8)))  float  v8f;

#define T_TOK 8192
#define HDIM  512
#define NEXP  8
#define IMOE  1024
#define ISH   2048
#define MTILE 64

union BFrag { uint4 q[2]; v16bf v; };

__device__ __forceinline__ v8f wmma_bf16(v16bf a, v16bf b, v8f c) {
  // D = A(16x32 bf16) x B(32x16 bf16) + C(16x16 f32)
  return __builtin_amdgcn_wmma_f32_16x16x32_bf16(false, a, false, b,
                                                 (short)0, c, false, false);
}

// A fragment (ISA 7.12.2, 16-bit A 16x32): lane<16 holds row=lane, K {k0..k0+7,
// k0+16..k0+23}; lane>=16 holds row=lane-16, K {k0+8..,k0+24..}. Row-major src.
__device__ __forceinline__ v16bf load_frag_a(const ushort_t* base, int stride,
                                             int row, int k0, int halfsel) {
  const ushort_t* p = base + row * stride + k0 + halfsel * 8;
  BFrag f;
  f.q[0] = *(const uint4*)(p);
  f.q[1] = *(const uint4*)(p + 16);
  return f.v;
}

// B fragment (K x 16): lane<16 = column lane with K {k0..k0+7, k0+16..k0+23};
// lane>=16 = column lane-16 with K {k0+8..,k0+24..}. B[k][n] = W[n][k] => 2x16 bytes
// from row n of a row-major weight matrix.
__device__ __forceinline__ v16bf load_frag_b(const ushort_t* base, long row,
                                             int ldw, int k0, int halfsel) {
  const ushort_t* p = base + row * (long)ldw + k0 + halfsel * 8;
  BFrag f;
  f.q[0] = *(const uint4*)(p);
  f.q[1] = *(const uint4*)(p + 16);
  return f.v;
}

__device__ __forceinline__ ushort_t f2bf(float f) {
  __hip_bfloat16 h = __float2bfloat16(f);
  return *reinterpret_cast<ushort_t*>(&h);
}

// ---------------------------------------------------------------------------
// fp32 -> bf16 bulk conversion (one-time; weights then live in L2 as bf16)
// ---------------------------------------------------------------------------
__global__ void cvt_bf16_kernel(const float* __restrict__ in,
                                ushort_t* __restrict__ out, int n) {
  int i = blockIdx.x * blockDim.x + threadIdx.x;
  int stride = gridDim.x * blockDim.x;
  for (; i < n; i += stride) out[i] = f2bf(in[i]);
}

// ---------------------------------------------------------------------------
// Router: 8 lanes per token. Writes logits (fp32, exact) to d_out tail,
// combine weights [T,8] (0 for unselected), and sigmoid shared gate [T].
// ---------------------------------------------------------------------------
__global__ __launch_bounds__(256) void router_kernel(
    const float* __restrict__ X, const float* __restrict__ Gw,
    const float* __restrict__ SGw, float* __restrict__ logits_out,
    float* __restrict__ combine, float* __restrict__ sgate) {
  int tid = threadIdx.x;
  int token = blockIdx.x * 32 + (tid >> 3);
  int e = tid & 7;
  const float* xr = X + (size_t)token * HDIM;
  const float4* x4 = (const float4*)xr;
  const float4* g4 = (const float4*)(Gw + e * HDIM);
  float acc = 0.f;
  for (int k = 0; k < HDIM / 4; ++k) {
    float4 a = x4[k], b = g4[k];
    acc += a.x * b.x + a.y * b.y + a.z * b.z + a.w * b.w;
  }
  logits_out[token * NEXP + e] = acc;

  // softmax over the 8 lanes of this token (wave32 xor-shuffles stay in group)
  float m = acc;
  m = fmaxf(m, __shfl_xor(m, 1, 32));
  m = fmaxf(m, __shfl_xor(m, 2, 32));
  m = fmaxf(m, __shfl_xor(m, 4, 32));
  float p = __expf(acc - m);
  float s = p;
  s += __shfl_xor(s, 1, 32);
  s += __shfl_xor(s, 2, 32);
  s += __shfl_xor(s, 4, 32);
  p /= s;

  // rank: count strictly-greater (ties broken by smaller expert index, which
  // matches jax.lax.top_k ordering). Exactly 2 lanes get cnt < 2.
  int gbase = (tid & 31) & ~7;
  int cnt = 0;
#pragma unroll
  for (int j = 0; j < 8; ++j) {
    float pj = __shfl(p, gbase + j, 32);
    cnt += (pj > p) || (pj == p && j < e);
  }
  combine[token * NEXP + e] = (cnt < 2) ? p : 0.f;

  // shared-expert sigmoid gate: 8-lane split dot product
  float g = 0.f;
  for (int k = e; k < HDIM; k += 8) g += xr[k] * SGw[k];
  g += __shfl_xor(g, 1, 32);
  g += __shfl_xor(g, 2, 32);
  g += __shfl_xor(g, 4, 32);
  if (e == 0) sgate[token] = 1.f / (1.f + __expf(-g));
}

// ---------------------------------------------------------------------------
// Deterministic per-expert compaction: one block per expert, LDS prefix scan.
// List entry = token*2 + slot, slot = #selected experts with smaller index.
// ---------------------------------------------------------------------------
__global__ __launch_bounds__(1024) void compact_kernel(
    const float* __restrict__ combine, int* __restrict__ lists,
    int* __restrict__ counts) {
  __shared__ int ps[1024];
  int e = blockIdx.x;
  int tid = threadIdx.x;
  const int PT = T_TOK / 1024;  // 8 tokens per thread
  int base = tid * PT;
  int flag[PT];
  int cnt = 0;
#pragma unroll
  for (int j = 0; j < PT; ++j) {
    flag[j] = combine[(size_t)(base + j) * NEXP + e] > 0.f;
    cnt += flag[j];
  }
  ps[tid] = cnt;
  __syncthreads();
  for (int off = 1; off < 1024; off <<= 1) {
    int add = (tid >= off) ? ps[tid - off] : 0;
    __syncthreads();
    ps[tid] += add;
    __syncthreads();
  }
  int pos = ps[tid] - cnt;
#pragma unroll
  for (int j = 0; j < PT; ++j) {
    if (flag[j]) {
      int t = base + j;
      int slot = 0;
      for (int ee = 0; ee < e; ++ee)
        slot += (combine[(size_t)t * NEXP + ee] > 0.f);
      lists[e * T_TOK + pos] = t * 2 + slot;
      ++pos;
    }
  }
  if (tid == 1023) counts[e] = ps[1023];
}

// ---------------------------------------------------------------------------
// Fused SwiGLU FFN GEMM. Block = 256 threads (8 wave32), tile = 64 rows x 512.
// Per 64-wide I-chunk:
//   phase 1: h = silu(x@w1^T)*(x@w3^T). Wave w owns I-tile (w&3) for an
//            M-pair (w>>2): each B fragment feeds 2 WMMAs in-register.
//   phase 2: out(64x512) += h(64x64) @ w2chunk^T. Wave w owns cols w*64..+63:
//            16 persistent f32 C-tiles (128 acc VGPRs/lane).
// mode 0: routed expert, gathered rows, scale=combine, plain store.
// mode 1: shared expert, identity rows, scale=sgate, accumulate into Y[2t].
// ---------------------------------------------------------------------------
__global__ __launch_bounds__(256) void moe_gemm_kernel(
    const ushort_t* __restrict__ Xbf, const ushort_t* __restrict__ W1,
    const ushort_t* __restrict__ W3, const ushort_t* __restrict__ W2, int I,
    const int* __restrict__ lists, const int* __restrict__ counts,
    const float* __restrict__ scales, float* __restrict__ Y, int mode) {
  __shared__ __align__(16) ushort_t Xs[MTILE * 520];  // +8 pad kills 16-way
  __shared__ __align__(16) ushort_t Hs[MTILE * 72];   // LDS bank conflicts
  __shared__ int Stok[MTILE];
  __shared__ int Sdst[MTILE];
  __shared__ float Sscale[MTILE];

  const int tid = threadIdx.x;
  const int wid = tid >> 5;
  const int lane = tid & 31;
  const int e = blockIdx.y;
  const int tile = blockIdx.x;

  if (mode == 0) {
    if (tile * MTILE >= counts[e]) return;  // uniform exit: EXEC stays full
  }

  const ushort_t* W1e = W1 + (size_t)e * I * HDIM;
  const ushort_t* W3e = W3 + (size_t)e * I * HDIM;
  const ushort_t* W2e = W2 + (size_t)e * HDIM * I;

  if (tid < MTILE) {
    int idx = tile * MTILE + tid;
    int tok, dst;
    float sc;
    if (mode == 0) {
      int n = counts[e];
      if (idx < n) {
        int entry = lists[e * T_TOK + idx];
        tok = entry >> 1;
        dst = entry;
        sc = scales[tok * NEXP + e];
      } else {  // padded row -> dummy Y row, zero scale
        tok = 0;
        dst = 2 * T_TOK + tid;
        sc = 0.f;
      }
    } else {
      tok = idx;
      dst = 2 * idx;
      sc = scales[idx];
    }
    Stok[tid] = tok;
    Sdst[tid] = dst;
    Sscale[tid] = sc;
  }
  __syncthreads();

  // gather 64 token rows of bf16 x into LDS (4096 uint4 / 256 threads)
  for (int i = tid; i < MTILE * 64; i += 256) {
    int r = i >> 6, c = i & 63;
    const uint4* src = (const uint4*)(Xbf + (size_t)Stok[r] * HDIM) + c;
    *(uint4*)(Xs + r * 520 + c * 8) = *src;
  }
  __syncthreads();

  const int halfsel = lane >> 4;
  const int lrow = lane & 15;
  const int mtp = wid >> 2;   // h-phase M-pair (0..1) -> M tiles 2*mtp, 2*mtp+1
  const int it_h = wid & 3;   // h-phase I tile (0..3)

  const v8f vzero = {0.f, 0.f, 0.f, 0.f, 0.f, 0.f, 0.f, 0.f};
  v8f acc[16];
#pragma unroll
  for (int i = 0; i < 16; ++i) acc[i] = vzero;

  const int nchunk = I >> 6;
  const long brow = (long)(it_h * 16 + lrow);
  for (int ch = 0; ch < nchunk; ++ch) {
    const int i0 = ch << 6;
    // ---- phase 1: h1 = x@w1^T, h3 = x@w3^T (2 M-tiles per wave) ----
    v8f h1a[2] = {vzero, vzero};
    v8f h3a[2] = {vzero, vzero};
#pragma unroll 4
    for (int kk = 0; kk < 16; ++kk) {
      const int k0 = kk << 5;
      v16bf A0 = load_frag_a(Xs, 520, (mtp * 2 + 0) * 16 + lrow, k0, halfsel);
      v16bf A1 = load_frag_a(Xs, 520, (mtp * 2 + 1) * 16 + lrow, k0, halfsel);
      v16bf B1 = load_frag_b(W1e, i0 + brow, HDIM, k0, halfsel);
      v16bf B3 = load_frag_b(W3e, i0 + brow, HDIM, k0, halfsel);
      h1a[0] = wmma_bf16(A0, B1, h1a[0]);
      h1a[1] = wmma_bf16(A1, B1, h1a[1]);
      h3a[0] = wmma_bf16(A0, B3, h3a[0]);
      h3a[1] = wmma_bf16(A1, B3, h3a[1]);
    }
    // silu(h1)*h3 via v_exp + v_rcp (avoid full IEEE divide chain), cvt bf16,
    // stage in LDS (C layout: value j of lane -> M = 8*halfsel + j, N = lrow)
#pragma unroll
    for (int sub = 0; sub < 2; ++sub) {
#pragma unroll
      for (int j = 0; j < 8; ++j) {
        float v1 = h1a[sub][j], v3 = h3a[sub][j];
        float sig = __builtin_amdgcn_rcpf(1.f + __expf(-v1));
        float h = v1 * sig * v3;
        int m = (mtp * 2 + sub) * 16 + halfsel * 8 + j;
        int c = it_h * 16 + lrow;
        Hs[m * 72 + c] = f2bf(h);
      }
    }
    __syncthreads();
    // ---- phase 2: out(64x512) += h(64x64) @ w2chunk^T, wave = 64 cols ----
#pragma unroll
    for (int ks = 0; ks < 2; ++ks) {
      const int k0 = ks << 5;
      v16bf A[4];
#pragma unroll
      for (int mt = 0; mt < 4; ++mt)
        A[mt] = load_frag_a(Hs, 72, mt * 16 + lrow, k0, halfsel);
#pragma unroll
      for (int nt = 0; nt < 4; ++nt) {
        v16bf B = load_frag_b(W2e, (long)(wid * 64 + nt * 16 + lrow), I,
                              i0 + k0, halfsel);
#pragma unroll
        for (int mt = 0; mt < 4; ++mt)
          acc[mt * 4 + nt] = wmma_bf16(A[mt], B, acc[mt * 4 + nt]);
      }
    }
    __syncthreads();
  }

  // epilogue: scale rows by combine/gate weight, scatter to unique Y rows
#pragma unroll
  for (int mt = 0; mt < 4; ++mt) {
#pragma unroll
    for (int nt = 0; nt < 4; ++nt) {
      v8f a = acc[mt * 4 + nt];
      int col = wid * 64 + nt * 16 + lrow;
#pragma unroll
      for (int j = 0; j < 8; ++j) {
        int m = mt * 16 + halfsel * 8 + j;
        float val = Sscale[m] * a[j];
        size_t off = (size_t)Sdst[m] * HDIM + col;
        if (mode == 0)
          Y[off] = val;
        else
          Y[off] += val;  // stream-ordered after routed pass: no race
      }
    }
  }
}

// ---------------------------------------------------------------------------
// out[t] = y[2t] + y[2t+1]  (slot0 row already holds routed0 + gate*shared)
// ---------------------------------------------------------------------------
__global__ void final_kernel(const float* __restrict__ Y,
                             float* __restrict__ out) {
  int i = blockIdx.x * blockDim.x + threadIdx.x;  // float4 index
  int t = i >> 7;
  int c4 = i & 127;
  const float4 y0 = ((const float4*)(Y + (size_t)(2 * t) * HDIM))[c4];
  const float4 y1 = ((const float4*)(Y + (size_t)(2 * t + 1) * HDIM))[c4];
  float4 o = make_float4(y0.x + y1.x, y0.y + y1.y, y0.z + y1.z, y0.w + y1.w);
  ((float4*)out)[i] = o;
}

// ---------------------------------------------------------------------------
extern "C" void kernel_launch(void* const* d_in, const int* in_sizes, int n_in,
                              void* d_out, int out_size, void* d_ws,
                              size_t ws_size, hipStream_t stream) {
  const float* X = (const float*)d_in[0];
  const float* Gw = (const float*)d_in[1];
  const float* EW1 = (const float*)d_in[2];
  const float* EW2 = (const float*)d_in[3];
  const float* EW3 = (const float*)d_in[4];
  const float* SW1 = (const float*)d_in[5];
  const float* SW2 = (const float*)d_in[6];
  const float* SW3 = (const float*)d_in[7];
  const float* SGw = (const float*)d_in[8];

  char* ws = (char*)d_ws;
  size_t off = 0;
  auto alloc = [&](size_t bytes) -> char* {
    char* p = ws + off;
    off = (off + bytes + 255) & ~(size_t)255;
    return p;
  };
  ushort_t* xbf = (ushort_t*)alloc((size_t)T_TOK * HDIM * 2);
  ushort_t* ew1 = (ushort_t*)alloc((size_t)NEXP * IMOE * HDIM * 2);
  ushort_t* ew3 = (ushort_t*)alloc((size_t)NEXP * IMOE * HDIM * 2);
  ushort_t* ew2 = (ushort_t*)alloc((size_t)NEXP * HDIM * IMOE * 2);
  ushort_t* sw1 = (ushort_t*)alloc((size_t)ISH * HDIM * 2);
  ushort_t* sw3 = (ushort_t*)alloc((size_t)ISH * HDIM * 2);
  ushort_t* sw2 = (ushort_t*)alloc((size_t)HDIM * ISH * 2);
  float* combine = (float*)alloc((size_t)T_TOK * NEXP * 4);
  float* sgate = (float*)alloc((size_t)T_TOK * 4);
  int* counts = (int*)alloc(256);
  int* lists = (int*)alloc((size_t)NEXP * T_TOK * 4);
  float* Ybuf = (float*)alloc((size_t)(2 * T_TOK + MTILE) * HDIM * 4);

  float* out = (float*)d_out;
  float* logits = out + (size_t)T_TOK * HDIM;

  // 1) one-time bf16 conversion (weights then stay resident in 192MB L2)
  cvt_bf16_kernel<<<2048, 256, 0, stream>>>(X, xbf, T_TOK * HDIM);
  cvt_bf16_kernel<<<2048, 256, 0, stream>>>(EW1, ew1, NEXP * IMOE * HDIM);
  cvt_bf16_kernel<<<2048, 256, 0, stream>>>(EW3, ew3, NEXP * IMOE * HDIM);
  cvt_bf16_kernel<<<2048, 256, 0, stream>>>(EW2, ew2, NEXP * HDIM * IMOE);
  cvt_bf16_kernel<<<1024, 256, 0, stream>>>(SW1, sw1, ISH * HDIM);
  cvt_bf16_kernel<<<1024, 256, 0, stream>>>(SW3, sw3, ISH * HDIM);
  cvt_bf16_kernel<<<1024, 256, 0, stream>>>(SW2, sw2, HDIM * ISH);

  // 2) router (logits to d_out tail, combine weights, sigmoid shared gate)
  router_kernel<<<T_TOK / 32, 256, 0, stream>>>(X, Gw, SGw, logits, combine,
                                                sgate);

  // 3) deterministic per-expert token compaction
  compact_kernel<<<NEXP, 1024, 0, stream>>>(combine, lists, counts);

  // 4) routed experts: sparse top-2 dispatch (fixed worst-case grid, blocks
  //    beyond ceil(count/64) exit immediately)
  moe_gemm_kernel<<<dim3(T_TOK / MTILE, NEXP), 256, 0, stream>>>(
      xbf, ew1, ew3, ew2, IMOE, lists, counts, combine, Ybuf, 0);

  // 5) shared expert, gate-scaled, accumulated into slot-0 rows
  moe_gemm_kernel<<<dim3(T_TOK / MTILE, 1), 256, 0, stream>>>(
      xbf, sw1, sw3, sw2, ISH, nullptr, nullptr, sgate, Ybuf, 1);

  // 6) final combine out[t] = y[2t] + y[2t+1]
  final_kernel<<<(T_TOK * HDIM / 4) / 256, 256, 0, stream>>>(Ybuf, out);
}
